// Attention_61710090109381
// MI455X (gfx1250) — 16-bit and 8-bit matrix operand order repaired and checked by reading, compiled, not run on hardware
//
#include <hip/hip_runtime.h>
#include <stdint.h>

#define H_    32
#define KVH_  8
#define D_    128
#define BS_   16
#define R_    4
#define S_    4
#define LSEQ_ 1024
#define T_    4096
#define BD_   32
#define M_    128
#define NEGVAL  (-1e10f)
#define SCALE_  0.08838834764831845f

typedef __attribute__((ext_vector_type(16))) _Float16 v16h;
typedef __attribute__((ext_vector_type(8)))  _Float16 v8h;
typedef __attribute__((ext_vector_type(8)))  float    v8f;
typedef __attribute__((ext_vector_type(4)))  float    v4f;
typedef __attribute__((ext_vector_type(4)))  int      v4i;
typedef __attribute__((ext_vector_type(8)))  int      v8i;

#define WMMA_F16(A, B, C) \
  __builtin_amdgcn_wmma_f32_16x16x32_f16(false, (A), false, (B), (short)0, (C), false, false)
#define WMMA_IU8(A, B, C) \
  __builtin_amdgcn_wmma_i32_16x16x64_iu8(true, (A), true, (B), (C), false, false)

#define LDS_OFF(p) \
  ((uint32_t)(uintptr_t)(__attribute__((address_space(3))) const void*)(p))

static __device__ __forceinline__ void async_copy_b128(uint32_t lds_off, const void* gptr) {
  asm volatile("global_load_async_to_lds_b128 %0, %1, off"
               :: "v"(lds_off), "v"(gptr) : "memory");
}
static __device__ __forceinline__ void wait_async_le8() {
  asm volatile("s_wait_asynccnt 0x8" ::: "memory");
}
static __device__ __forceinline__ void wait_async_0() {
  asm volatile("s_wait_asynccnt 0x0" ::: "memory");
}

static __device__ __forceinline__ v16h mkA_f32(const float* plo, const float* phi, float sc) {
  v4f a0 = *(const v4f*)plo,  a1 = *(const v4f*)(plo + 4);
  v4f b0 = *(const v4f*)phi,  b1 = *(const v4f*)(phi + 4);
  v16h r;
#pragma unroll
  for (int i = 0; i < 4; i++) {
    r[i]      = (_Float16)(a0[i] * sc);
    r[4 + i]  = (_Float16)(a1[i] * sc);
    r[8 + i]  = (_Float16)(b0[i] * sc);
    r[12 + i] = (_Float16)(b1[i] * sc);
  }
  return r;
}

static __device__ __forceinline__ v16h mkB_f32(const float* p, float sc) {
  v4f x0 = ((const v4f*)p)[0], x1 = ((const v4f*)p)[1];
  v4f x2 = ((const v4f*)p)[2], x3 = ((const v4f*)p)[3];
  v16h r;
#pragma unroll
  for (int i = 0; i < 4; i++) {
    r[i]      = (_Float16)(x0[i] * sc);
    r[4 + i]  = (_Float16)(x1[i] * sc);
    r[8 + i]  = (_Float16)(x2[i] * sc);
    r[12 + i] = (_Float16)(x3[i] * sc);
  }
  return r;
}

static __device__ __forceinline__ v8f vzero8() {
  v8f z;
#pragma unroll
  for (int i = 0; i < 8; i++) z[i] = 0.f;
  return z;
}
static __device__ __forceinline__ v8i vzeroi8() {
  v8i z;
#pragma unroll
  for (int i = 0; i < 8; i++) z[i] = 0;
  return z;
}
static __device__ __forceinline__ int pack4(int b0, int b1, int b2, int b3) {
  return (b0 & 0xff) | ((b1 & 0xff) << 8) | ((b2 & 0xff) << 16) | ((b3 & 0xff) << 24);
}
union HalfPack { _Float16 h[2]; uint32_t u; };

__global__ __launch_bounds__(128, 1) void prefill_attn(
    const float* __restrict__ q, const float* __restrict__ k,
    const float* __restrict__ v, float* __restrict__ out) {
  __shared__ float    Vraw[2][32 * D_];
  __shared__ _Float16 Vs[D_ * 32];
  __shared__ _Float16 Ps[4][16 * 32];

  const int qt = blockIdx.x, h = blockIdx.y, s = blockIdx.z;
  const int g = h >> 2;
  const int tid = threadIdx.x, wave = tid >> 5, lane = tid & 31;
  const int n = lane & 15, hi = lane >> 4;
  const int seq = s * LSEQ_;
  const int qrow0 = qt * 64 + wave * 16;

  v16h qa[4];
  {
    const float* qp = q + ((size_t)(seq + qrow0 + n) * H_ + h) * D_;
#pragma unroll
    for (int c = 0; c < 4; c++)
      qa[c] = mkA_f32(qp + c * 32 + hi * 16, qp + c * 32 + hi * 16 + 8, SCALE_);
  }

  float m_r[8], l_r[8];
  v8f o[8];
#pragma unroll
  for (int r = 0; r < 8; r++) { m_r[r] = -1e30f; l_r[r] = 0.f; }
#pragma unroll
  for (int t = 0; t < 8; t++) o[t] = vzero8();

  const int ktmax = (qt + 1) * 2;

  auto issue_vtile = [&](int kt) {
#pragma unroll
    for (int i = 0; i < 8; i++) {
      const int ci  = tid + i * 128;
      const int key = ci >> 5;
      const int off = (ci & 31) * 4;
      const float* gp = v + ((size_t)(seq + kt * 32 + key) * KVH_ + g) * D_ + off;
      async_copy_b128(LDS_OFF(&Vraw[kt & 1][ci * 4]), gp);
    }
  };

  issue_vtile(0);

  for (int kt = 0; kt < ktmax; ++kt) {
    if (kt + 1 < ktmax) { issue_vtile(kt + 1); wait_async_le8(); }
    else                { wait_async_0(); }
    __syncthreads();

    {
      const float* src = &Vraw[kt & 1][0];
      const int jp = tid & 15;
      const int dblk = tid >> 4;
      const float* r0 = src + (2 * jp) * D_ + dblk * 16;
      const float* r1 = src + (2 * jp + 1) * D_ + dblk * 16;
#pragma unroll
      for (int c4 = 0; c4 < 4; c4++) {
        v4f a = *(const v4f*)(r0 + c4 * 4);
        v4f b = *(const v4f*)(r1 + c4 * 4);
#pragma unroll
        for (int e = 0; e < 4; e++) {
          HalfPack hp;
          hp.h[0] = (_Float16)a[e];
          hp.h[1] = (_Float16)b[e];
          *(uint32_t*)&Vs[(dblk * 16 + c4 * 4 + e) * 32 + 2 * jp] = hp.u;
        }
      }
    }
    __syncthreads();

    const int col0 = kt * 32 + n, col1 = col0 + 16;
    const float* kp0 = k + ((size_t)(seq + col0) * KVH_ + g) * D_;
    const float* kp1 = k + ((size_t)(seq + col1) * KVH_ + g) * D_;
    v8f s0 = vzero8(), s1 = vzero8();
#pragma unroll
    for (int c = 0; c < 4; c++) {
      s0 = WMMA_F16(qa[c], mkB_f32(kp0 + c * 32 + hi * 16, 1.f), s0);
      s1 = WMMA_F16(qa[c], mkB_f32(kp1 + c * 32 + hi * 16, 1.f), s1);
    }

    float al[8], p0v[8], p1v[8];
#pragma unroll
    for (int r = 0; r < 8; r++) {
      const int row = qrow0 + r + 8 * hi;
      float a = (col0 <= row) ? s0[r] : NEGVAL;
      float b = (col1 <= row) ? s1[r] : NEGVAL;
      float mx = fmaxf(a, b);
#pragma unroll
      for (int off = 1; off < 16; off <<= 1) mx = fmaxf(mx, __shfl_xor(mx, off, 32));
      const float mn = fmaxf(m_r[r], mx);
      al[r] = __expf(m_r[r] - mn);
      m_r[r] = mn;
      p0v[r] = __expf(a - mn);
      p1v[r] = __expf(b - mn);
      float rs = p0v[r] + p1v[r];
#pragma unroll
      for (int off = 1; off < 16; off <<= 1) rs += __shfl_xor(rs, off, 32);
      l_r[r] = l_r[r] * al[r] + rs;
    }
#pragma unroll
    for (int t = 0; t < 8; t++)
#pragma unroll
      for (int r = 0; r < 8; r++) o[t][r] *= al[r];

    _Float16* pw = &Ps[wave][0];
#pragma unroll
    for (int r = 0; r < 8; r++) {
      pw[(r + 8 * hi) * 32 + n]      = (_Float16)p0v[r];
      pw[(r + 8 * hi) * 32 + 16 + n] = (_Float16)p1v[r];
    }
    v8h plo = *(const v8h*)(pw + n * 32 + hi * 16);
    v8h phv = *(const v8h*)(pw + n * 32 + hi * 16 + 8);
    v16h pa = __builtin_shufflevector(plo, phv, 0, 1, 2, 3, 4, 5, 6, 7,
                                      8, 9, 10, 11, 12, 13, 14, 15);

#pragma unroll
    for (int t = 0; t < 8; t++) {
      v8h vlo = *(const v8h*)(Vs + (t * 16 + n) * 32 + hi * 16);
      v8h vhi = *(const v8h*)(Vs + (t * 16 + n) * 32 + hi * 16 + 8);
      v16h vb = __builtin_shufflevector(vlo, vhi, 0, 1, 2, 3, 4, 5, 6, 7,
                                        8, 9, 10, 11, 12, 13, 14, 15);
      o[t] = WMMA_F16(pa, vb, o[t]);
    }
  }

#pragma unroll
  for (int r = 0; r < 8; r++) {
    const int row = qrow0 + r + 8 * hi;
    const float inv = 1.f / l_r[r];
    float* op = out + ((size_t)(seq + row) * H_ + h) * D_;
#pragma unroll
    for (int t = 0; t < 8; t++) op[t * 16 + n] = o[t][r] * inv;
  }
}

__global__ __launch_bounds__(128, 1) void decode_attn(
    const float* __restrict__ query, const int8_t* __restrict__ kc,
    const int8_t* __restrict__ vc, const float* __restrict__ ks,
    const float* __restrict__ vsc, const int* __restrict__ bt,
    const int* __restrict__ ctxl, float* __restrict__ out) {
  __shared__ _Float16 Vsd[4][D_ * 32];
  __shared__ _Float16 Ps[4][16 * 32];
  __shared__ float Om[4][4][D_];
  __shared__ float Ml[4][4][2];

  const int g = blockIdx.x, b = blockIdx.y;
  const int tid = threadIdx.x, wave = tid >> 5, lane = tid & 31;
  const int n = lane & 15, hi = lane >> 4;
  const int ctx = ctxl[b];
  const int ntiles = (ctx + 31) >> 5;
  const int* btrow = bt + b * M_;

  float qf[4][16];
  {
    const int mh = (n < 4) ? n : 0;
    const float qm = (n < 4) ? 1.f : 0.f;
    const float* qp = query + ((size_t)b * H_ + g * R_ + mh) * D_;
#pragma unroll
    for (int c = 0; c < 4; c++) {
      v4f a0 = *(const v4f*)(qp + c * 32 + hi * 16);
      v4f a1 = *(const v4f*)(qp + c * 32 + hi * 16 + 4);
      v4f b0 = *(const v4f*)(qp + c * 32 + hi * 16 + 8);
      v4f b1 = *(const v4f*)(qp + c * 32 + hi * 16 + 12);
#pragma unroll
      for (int i = 0; i < 4; i++) {
        qf[c][i]      = a0[i] * qm;
        qf[c][4 + i]  = a1[i] * qm;
        qf[c][8 + i]  = b0[i] * qm;
        qf[c][12 + i] = b1[i] * qm;
      }
    }
  }
  float amax = 0.f;
#pragma unroll
  for (int c = 0; c < 4; c++)
#pragma unroll
    for (int j = 0; j < 16; j++) amax = fmaxf(amax, fabsf(qf[c][j]));
#pragma unroll
  for (int off = 1; off < 32; off <<= 1) amax = fmaxf(amax, __shfl_xor(amax, off, 32));
  const float qstep = fmaxf(amax, 1e-20f) * (1.f / 127.f);
  const float qinv  = 127.f / fmaxf(amax, 1e-20f);
  v8i qa0, qa1;
#pragma unroll
  for (int w = 0; w < 4; w++) {
    qa0[w]     = pack4(__float2int_rn(qf[0][4 * w] * qinv), __float2int_rn(qf[0][4 * w + 1] * qinv),
                       __float2int_rn(qf[0][4 * w + 2] * qinv), __float2int_rn(qf[0][4 * w + 3] * qinv));
    qa0[4 + w] = pack4(__float2int_rn(qf[1][4 * w] * qinv), __float2int_rn(qf[1][4 * w + 1] * qinv),
                       __float2int_rn(qf[1][4 * w + 2] * qinv), __float2int_rn(qf[1][4 * w + 3] * qinv));
    qa1[w]     = pack4(__float2int_rn(qf[2][4 * w] * qinv), __float2int_rn(qf[2][4 * w + 1] * qinv),
                       __float2int_rn(qf[2][4 * w + 2] * qinv), __float2int_rn(qf[2][4 * w + 3] * qinv));
    qa1[4 + w] = pack4(__float2int_rn(qf[3][4 * w] * qinv), __float2int_rn(qf[3][4 * w + 1] * qinv),
                       __float2int_rn(qf[3][4 * w + 2] * qinv), __float2int_rn(qf[3][4 * w + 3] * qinv));
  }

  float m_r[8], l_r[8];
  v8f o[8];
#pragma unroll
  for (int r = 0; r < 8; r++) { m_r[r] = -1e30f; l_r[r] = 0.f; }
#pragma unroll
  for (int t = 0; t < 8; t++) o[t] = vzero8();

  _Float16* Vw = &Vsd[wave][0];
  _Float16* pw = &Ps[wave][0];

  for (int kt = wave; kt < ntiles; kt += 4) {
    {
      const int jp = lane & 15;
      const int dh = lane >> 4;
      const int k0g = kt * 32 + 2 * jp, k1g = k0g + 1;
      const int b0i = btrow[k0g >> 4], b1i = btrow[k1g >> 4];
      const size_t s0i = ((size_t)((b0i < 0) ? 0 : b0i) * BS_ + (k0g & 15)) * KVH_ + g;
      const size_t s1i = ((size_t)((b1i < 0) ? 0 : b1i) * BS_ + (k1g & 15)) * KVH_ + g;
      const float vs0 = vsc[s0i], vs1 = vsc[s1i];
      const int8_t* vp0 = vc + s0i * D_ + dh * 64;
      const int8_t* vp1 = vc + s1i * D_ + dh * 64;
#pragma unroll
      for (int c16 = 0; c16 < 4; c16++) {
        v4i w0 = *(const v4i*)(vp0 + c16 * 16);
        v4i w1 = *(const v4i*)(vp1 + c16 * 16);
#pragma unroll
        for (int bw = 0; bw < 4; bw++) {
#pragma unroll
          for (int e = 0; e < 4; e++) {
            HalfPack hp;
            hp.h[0] = (_Float16)((float)(int8_t)(w0[bw] >> (8 * e)) * vs0);
            hp.h[1] = (_Float16)((float)(int8_t)(w1[bw] >> (8 * e)) * vs1);
            const int d = dh * 64 + c16 * 16 + bw * 4 + e;
            *(uint32_t*)&Vw[d * 32 + 2 * jp] = hp.u;
          }
        }
      }
    }

    const int col0 = kt * 32 + n, col1 = col0 + 16;
    const int p0i = btrow[col0 >> 4], p1i = btrow[col1 >> 4];
    const size_t sl0 = ((size_t)((p0i < 0) ? 0 : p0i) * BS_ + (col0 & 15)) * KVH_ + g;
    const size_t sl1 = ((size_t)((p1i < 0) ? 0 : p1i) * BS_ + (col1 & 15)) * KVH_ + g;
    const int8_t* kp0 = kc + sl0 * D_;
    const int8_t* kp1 = kc + sl1 * D_;
    v8i acc0 = vzeroi8(), acc1 = vzeroi8();
#pragma unroll
    for (int halfd = 0; halfd < 2; halfd++) {
      const int base = halfd * 64;
      v8i kb0, kb1;
      v4i x0 = *(const v4i*)(kp0 + base + hi * 16);
      v4i x1 = *(const v4i*)(kp0 + base + 32 + hi * 16);
      v4i y0 = *(const v4i*)(kp1 + base + hi * 16);
      v4i y1 = *(const v4i*)(kp1 + base + 32 + hi * 16);
#pragma unroll
      for (int i = 0; i < 4; i++) { kb0[i] = x0[i]; kb0[4 + i] = x1[i];
                                    kb1[i] = y0[i]; kb1[4 + i] = y1[i]; }
      acc0 = WMMA_IU8((halfd == 0) ? qa0 : qa1, kb0, acc0);
      acc1 = WMMA_IU8((halfd == 0) ? qa0 : qa1, kb1, acc1);
    }
    const float dq0 = qstep * SCALE_ * ks[sl0];
    const float dq1 = qstep * SCALE_ * ks[sl1];
    const bool ok0 = (col0 < ctx) && (p0i >= 0);
    const bool ok1 = (col1 < ctx) && (p1i >= 0);

    float al[8], p0v[8], p1v[8];
#pragma unroll
    for (int r = 0; r < 8; r++) {
      float a  = ok0 ? ((float)acc0[r] * dq0) : NEGVAL;
      float bb = ok1 ? ((float)acc1[r] * dq1) : NEGVAL;
      float mx = fmaxf(a, bb);
#pragma unroll
      for (int off = 1; off < 16; off <<= 1) mx = fmaxf(mx, __shfl_xor(mx, off, 32));
      const float mn = fmaxf(m_r[r], mx);
      al[r] = __expf(m_r[r] - mn);
      m_r[r] = mn;
      p0v[r] = __expf(a - mn);
      p1v[r] = __expf(bb - mn);
      float rs = p0v[r] + p1v[r];
#pragma unroll
      for (int off = 1; off < 16; off <<= 1) rs += __shfl_xor(rs, off, 32);
      l_r[r] = l_r[r] * al[r] + rs;
    }
#pragma unroll
    for (int t = 0; t < 8; t++)
#pragma unroll
      for (int r = 0; r < 8; r++) o[t][r] *= al[r];

#pragma unroll
    for (int r = 0; r < 8; r++) {
      pw[(r + 8 * hi) * 32 + n]      = (_Float16)p0v[r];
      pw[(r + 8 * hi) * 32 + 16 + n] = (_Float16)p1v[r];
    }
    v8h plo = *(const v8h*)(pw + n * 32 + hi * 16);
    v8h phv = *(const v8h*)(pw + n * 32 + hi * 16 + 8);
    v16h pa = __builtin_shufflevector(plo, phv, 0, 1, 2, 3, 4, 5, 6, 7,
                                      8, 9, 10, 11, 12, 13, 14, 15);

#pragma unroll
    for (int t = 0; t < 8; t++) {
      v8h vlo = *(const v8h*)(Vw + (t * 16 + n) * 32 + hi * 16);
      v8h vhi = *(const v8h*)(Vw + (t * 16 + n) * 32 + hi * 16 + 8);
      v16h vb = __builtin_shufflevector(vlo, vhi, 0, 1, 2, 3, 4, 5, 6, 7,
                                        8, 9, 10, 11, 12, 13, 14, 15);
      o[t] = WMMA_F16(pa, vb, o[t]);
    }
  }

  if (hi == 0) {
#pragma unroll
    for (int r = 0; r < 4; r++)
#pragma unroll
      for (int t = 0; t < 8; t++) Om[wave][r][t * 16 + n] = o[t][r];
    if (lane == 0) {
#pragma unroll
      for (int r = 0; r < 4; r++) { Ml[wave][r][0] = m_r[r]; Ml[wave][r][1] = l_r[r]; }
    }
  }
  __syncthreads();

  {
    const int r = tid >> 5;
    const int dd = tid & 31;
    float M = Ml[0][r][0];
#pragma unroll
    for (int w = 1; w < 4; w++) M = fmaxf(M, Ml[w][r][0]);
    float ew[4]; float L = 0.f;
#pragma unroll
    for (int w = 0; w < 4; w++) { ew[w] = __expf(Ml[w][r][0] - M); L += Ml[w][r][1] * ew[w]; }
    float* op = out + ((size_t)b * H_ + g * R_ + r) * D_;
#pragma unroll
    for (int c = 0; c < 4; c++) {
      const int d = dd + c * 32;
      float acc = 0.f;
#pragma unroll
      for (int w = 0; w < 4; w++) acc += Om[w][r][d] * ew[w];
      op[d] = acc / L;
    }
  }
}

extern "C" void kernel_launch(void* const* d_in, const int* in_sizes, int n_in,
                              void* d_out, int out_size, void* d_ws, size_t ws_size,
                              hipStream_t stream) {
  const float*   q       = (const float*)d_in[0];
  const float*   k       = (const float*)d_in[1];
  const float*   v       = (const float*)d_in[2];
  const float*   query   = (const float*)d_in[4];
  const int8_t*  k_cache = (const int8_t*)d_in[5];
  const int8_t*  v_cache = (const int8_t*)d_in[6];
  const float*   k_scale = (const float*)d_in[7];
  const float*   v_scale = (const float*)d_in[8];
  const int*     btab    = (const int*)d_in[9];
  const int*     ctxl    = (const int*)d_in[10];
  float* out = (float*)d_out;

  dim3 gp(LSEQ_ / 64, H_, S_);
  prefill_attn<<<gp, 128, 0, stream>>>(q, k, v, out);

  dim3 gd(KVH_, BD_);
  decode_attn<<<gd, 128, 0, stream>>>(query, k_cache, v_cache, k_scale, v_scale,
                                      btab, ctxl, out + (size_t)T_ * H_ * D_);
}
